// LSTM_47433618817195
// MI455X (gfx1250) — matrix operand order repaired and checked by reading, not compiled, not run on hardware
//
#include <hip/hip_runtime.h>
#include <hip/hip_bf16.h>
#include <stdint.h>

// ---------------- problem constants ----------------
#define IDIM    1024
#define HID     2048
#define FOURH   8192
#define TSEQ    4096
#define NBLK_REC 128          // persistent blocks; one j-tile (16 hidden units) each
#define COL_HALFS 2056        // 2048 halfs + 8 halfs (16B) pad per LDS weight column
// LDS: 64 weight cols + h(2048 bf16) + 64-float gate-exchange
#define REC_SMEM_BYTES (64 * COL_HALFS * 2 + HID * 2 + 64 * 4)

typedef __attribute__((ext_vector_type(16))) __bf16   bf16x16;
typedef __attribute__((ext_vector_type(8)))  float    f32x8;
typedef __attribute__((ext_vector_type(4)))  uint32_t u32x4;

union BV { bf16x16 v; u32x4 u[2]; };

__device__ __forceinline__ float sigmf(float x) { return 1.0f / (1.0f + __expf(-x)); }

__device__ __forceinline__ uint16_t f2bf(float f) {   // round-to-nearest-even
  uint32_t u = __float_as_uint(f);
  uint32_t r = u + 0x7FFFu + ((u >> 16) & 1u);
  return (uint16_t)(r >> 16);
}

// ---- CDNA5 async memory->LDS copy (CDNA5 ISA, ASYNCcnt) ----
// LDS operand = low 32 bits of the generic/shared pointer (LDS aperture
// truncation, ISA §10.2): ds-space addresses are addr[31:0].
__device__ __forceinline__ void async_b128(void* lds_ptr, const void* gptr) {
  asm volatile("global_load_async_to_lds_b128 %0, %1, off"
               :: "v"((uint32_t)(uintptr_t)lds_ptr),
                  "v"((uint64_t)(uintptr_t)gptr)
               : "memory");
}
__device__ __forceinline__ void wait_async0() {
  asm volatile("s_wait_asynccnt 0x0" ::: "memory");
}

// ---------------- init: zero h broadcast buffers + barrier ----------------
__global__ void k_init(uint16_t* hb, float* hf, unsigned int* bar) {
  int i = blockIdx.x * 256 + threadIdx.x;
  if (i < HID) { hb[i] = 0; hf[i] = 0.0f; }
  if (i == 0) bar[0] = 0u;
}

// ---------------- f32 -> bf16 bulk convert ----------------
__global__ void k_cvt(const float* __restrict__ src, uint16_t* __restrict__ dst, int n) {
  int i = blockIdx.x * 256 + threadIdx.x;
  if (i < n) dst[i] = f2bf(src[i]);
}

// ---------------- bias combine: b_ih + b_hh ----------------
__global__ void k_bias(const float* __restrict__ bih, const float* __restrict__ bhh,
                       float* __restrict__ bias) {
  int i = blockIdx.x * 256 + threadIdx.x;
  if (i < FOURH) bias[i] = bih[i] + bhh[i];
}

// ---------------- phase 1: xg = x @ W_ih^T + (b_ih+b_hh), bf16 WMMA ----------------
// one wave per 16x16 output tile; dual accumulators to split the D->C chain.
__global__ __launch_bounds__(256) void k_xg_gemm(const uint16_t* __restrict__ xb,
                                                 const uint16_t* __restrict__ wb,
                                                 const float* __restrict__ bias,
                                                 float* __restrict__ xg) {
  const int wave = threadIdx.x >> 5;
  const int lane = threadIdx.x & 31;
  const int ln = lane & 15;          // column within tile / A row within tile
  const int lh = lane >> 4;          // half-wave selects K sub-chunk
  const int tileIdx = blockIdx.x * 8 + wave;       // 256*512 tiles
  const int tcol = tileIdx & 511;                  // over 8192 gates
  const int trow = tileIdx >> 9;                   // over 4096 timesteps
  const int n0 = tcol * 16, t0 = trow * 16;

  const uint16_t* arow = xb + (size_t)(t0 + ln) * IDIM;
  const uint16_t* bcol = wb + (size_t)(n0 + ln) * IDIM;

  f32x8 acc0 = {}, acc1 = {};
#pragma unroll 2
  for (int k0 = 0; k0 < IDIM; k0 += 64) {
    BV a0, b0, a1, b1;
    a0.u[0] = *(const u32x4*)(arow + k0 + lh * 8);
    a0.u[1] = *(const u32x4*)(arow + k0 + 16 + lh * 8);
    b0.u[0] = *(const u32x4*)(bcol + k0 + lh * 8);
    b0.u[1] = *(const u32x4*)(bcol + k0 + 16 + lh * 8);
    a1.u[0] = *(const u32x4*)(arow + k0 + 32 + lh * 8);
    a1.u[1] = *(const u32x4*)(arow + k0 + 48 + lh * 8);
    b1.u[0] = *(const u32x4*)(bcol + k0 + 32 + lh * 8);
    b1.u[1] = *(const u32x4*)(bcol + k0 + 48 + lh * 8);
    acc0 = __builtin_amdgcn_wmma_f32_16x16x32_bf16(false, a0.v, false, b0.v,
                                                   (short)0, acc0, false, false);
    acc1 = __builtin_amdgcn_wmma_f32_16x16x32_bf16(false, a1.v, false, b1.v,
                                                   (short)0, acc1, false, false);
  }
  const float bv = bias[n0 + ln];
#pragma unroll
  for (int r = 0; r < 8; ++r)
    xg[(size_t)(t0 + r + 8 * lh) * FOURH + n0 + ln] = acc0[r] + acc1[r] + bv;
}

// ---------------- device-wide barrier (monotonic counter, agent scope) ----------------
__device__ __forceinline__ void grid_sync(unsigned int* bar, unsigned int target) {
  __syncthreads();
  if (threadIdx.x == 0) {
    __hip_atomic_fetch_add(bar, 1u, __ATOMIC_ACQ_REL, __HIP_MEMORY_SCOPE_AGENT);
    while (__hip_atomic_load(bar, __ATOMIC_ACQUIRE, __HIP_MEMORY_SCOPE_AGENT) < target)
      __builtin_amdgcn_s_sleep(2);
  }
  __syncthreads();   // acquire by t0 + block barrier orders everyone's reloads
}

// ---------------- phase 2: persistent LSTM recurrence ----------------
// block b owns hidden units j in [16b,16b+16): gate rows {g*2048+16b..+16}, g=0..3.
// W_hh slice (64 rows x 2048 k, bf16) staged to LDS once via async copies
// (257 KB of the 320 KB/WGP). Per step: h (4 KB bf16) global->LDS (async),
// GEMV via bf16 WMMA with h replicated over all 16 A rows (every lane's acc[0]
// = dot(h, W_col)), gate math on 16 lanes, one device barrier.
__global__ __launch_bounds__(128, 1) void k_lstm_rec(const float* __restrict__ xg,
                                                     const uint16_t* __restrict__ whh,
                                                     uint16_t* __restrict__ h_bf16,
                                                     float* __restrict__ h_f32,
                                                     unsigned int* __restrict__ bar) {
  extern __shared__ char smem[];
  uint16_t* sW = (uint16_t*)smem;                 // 64 cols * COL_HALFS
  uint16_t* sh = sW + 64 * COL_HALFS;             // h, 2048 halfs
  float*    gx = (float*)(sh + HID);              // 4 gates * 16 values

  const int tid  = threadIdx.x;
  const int wave = tid >> 5;                      // gate g = wave
  const int lane = tid & 31;
  const int ln   = lane & 15;
  const int lh   = lane >> 4;
  const int jb   = blockIdx.x;                    // j-tile

  // ---- prologue: stage W_hh slice into LDS (async, column-major per gate col)
  {
    const int col  = tid >> 1;                    // 0..63  (g*16 + n)
    const int part = tid & 1;                     // low/high 1024 k
    const int g = col >> 4, n = col & 15;
    const uint16_t* gsrc = whh + (size_t)(g * HID + 16 * jb + n) * HID + part * 1024;
    uint16_t* ldst = sW + (size_t)col * COL_HALFS + part * 1024;
#pragma unroll 4
    for (int i = 0; i < 128; ++i)
      async_b128(ldst + i * 8, gsrc + i * 8);
    wait_async0();
  }
  __syncthreads();

  float c_state = 0.0f;                           // live in lanes tid<16 only
  const uint16_t* colbase = sW + (size_t)(wave * 16 + ln) * COL_HALFS;

  for (int t = 0; t < TSEQ; ++t) {
    // broadcast h into LDS (4 KB) without a VGPR round-trip
    async_b128(sh + tid * 8,        h_bf16 + tid * 8);
    async_b128(sh + 1024 + tid * 8, h_bf16 + 1024 + tid * 8);
    wait_async0();
    __syncthreads();

    // hoist the xg_t term so its L2 latency overlaps the WMMA chain
    const float xgv = xg[(size_t)t * FOURH + wave * HID + 16 * jb + ln];

    f32x8 acc0 = {}, acc1 = {};
#pragma unroll 4
    for (int k0 = 0; k0 < HID; k0 += 64) {
      BV a0, b0, a1, b1;
      a0.u[0] = *(const u32x4*)(sh + k0 + lh * 8);        // h replicated across rows
      a0.u[1] = *(const u32x4*)(sh + k0 + 16 + lh * 8);
      b0.u[0] = *(const u32x4*)(colbase + k0 + lh * 8);   // W col, same k order as A
      b0.u[1] = *(const u32x4*)(colbase + k0 + 16 + lh * 8);
      a1.u[0] = *(const u32x4*)(sh + k0 + 32 + lh * 8);
      a1.u[1] = *(const u32x4*)(sh + k0 + 48 + lh * 8);
      b1.u[0] = *(const u32x4*)(colbase + k0 + 32 + lh * 8);
      b1.u[1] = *(const u32x4*)(colbase + k0 + 48 + lh * 8);
      acc0 = __builtin_amdgcn_wmma_f32_16x16x32_bf16(false, a0.v, false, b0.v,
                                                     (short)0, acc0, false, false);
      acc1 = __builtin_amdgcn_wmma_f32_16x16x32_bf16(false, a1.v, false, b1.v,
                                                     (short)0, acc1, false, false);
    }
    // every lane's acc[0] holds dot(h, W_hh[row n0+ln]); add xg_t (+biases)
    const float g = acc0[0] + acc1[0] + xgv;
    if (lane < 16) gx[wave * 16 + ln] = g;
    __syncthreads();

    if (tid < 16) {
      const float iv = sigmf(gx[tid]);
      const float fv = sigmf(gx[16 + tid]);
      const float gv = tanhf(gx[32 + tid]);
      const float ov = sigmf(gx[48 + tid]);
      c_state = fv * c_state + iv * gv;
      const float hv = ov * tanhf(c_state);
      const int j = 16 * jb + tid;
      h_f32[j]  = hv;
      h_bf16[j] = f2bf(hv);
    }
    grid_sync(bar, (unsigned int)(t + 1) * NBLK_REC);
  }
}

// ---------------- phase 3: out = h_T . W_out + b_out ----------------
__global__ void k_final(const float* __restrict__ h, const float* __restrict__ wout,
                        const float* __restrict__ bout, float* __restrict__ out) {
  __shared__ float red[256];
  float s = 0.0f;
  for (int j = threadIdx.x; j < HID; j += 256) s += h[j] * wout[j];
  red[threadIdx.x] = s;
  __syncthreads();
  for (int w = 128; w > 0; w >>= 1) {
    if (threadIdx.x < w) red[threadIdx.x] += red[threadIdx.x + w];
    __syncthreads();
  }
  if (threadIdx.x == 0) out[0] = red[0] + bout[0];
}

// ---------------- launch ----------------
extern "C" void kernel_launch(void* const* d_in, const int* in_sizes, int n_in,
                              void* d_out, int out_size, void* d_ws, size_t ws_size,
                              hipStream_t stream) {
  (void)in_sizes; (void)n_in; (void)out_size; (void)ws_size;
  const float* x    = (const float*)d_in[0];   // (1,4096,1024)
  const float* Wih  = (const float*)d_in[1];   // (8192,1024)
  const float* Whh  = (const float*)d_in[2];   // (8192,2048)
  const float* bih  = (const float*)d_in[3];   // (8192,)
  const float* bhh  = (const float*)d_in[4];   // (8192,)
  const float* Wout = (const float*)d_in[5];   // (1,2048)
  const float* bout = (const float*)d_in[6];   // (1,)
  float* out = (float*)d_out;

  // workspace carve-up (~194 MB)
  char* p = (char*)d_ws;
  size_t off = 0;
  auto take = [&](size_t bytes) { char* q = p + off; off = (off + bytes + 255) & ~(size_t)255; return q; };
  float*    xg    = (float*)   take((size_t)TSEQ * FOURH * 4);
  uint16_t* xb    = (uint16_t*)take((size_t)TSEQ * IDIM * 2);
  uint16_t* wihb  = (uint16_t*)take((size_t)FOURH * IDIM * 2);
  uint16_t* whhb  = (uint16_t*)take((size_t)FOURH * HID * 2);
  float*    bias  = (float*)   take((size_t)FOURH * 4);
  uint16_t* hbf   = (uint16_t*)take((size_t)HID * 2);
  float*    hf32  = (float*)   take((size_t)HID * 4);
  unsigned int* bar = (unsigned int*)take(256);

  k_init<<<(HID + 255) / 256, 256, 0, stream>>>(hbf, hf32, bar);

  k_cvt<<<(TSEQ * IDIM + 255) / 256, 256, 0, stream>>>(x, xb, TSEQ * IDIM);
  k_cvt<<<(FOURH * IDIM + 255) / 256, 256, 0, stream>>>(Wih, wihb, FOURH * IDIM);
  k_cvt<<<(FOURH * HID + 255) / 256, 256, 0, stream>>>(Whh, whhb, FOURH * HID);
  k_bias<<<(FOURH + 255) / 256, 256, 0, stream>>>(bih, bhh, bias);

  // 256 t-tiles * 512 n-tiles, 8 waves (tiles) per 256-thread block
  k_xg_gemm<<<(256 * 512) / 8, 256, 0, stream>>>(xb, wihb, bias, xg);

  static_assert(REC_SMEM_BYTES <= 320 * 1024, "LDS over WGP budget");
  (void)hipFuncSetAttribute(reinterpret_cast<const void*>(k_lstm_rec),
                            hipFuncAttributeMaxDynamicSharedMemorySize, REC_SMEM_BYTES);
  k_lstm_rec<<<NBLK_REC, 128, REC_SMEM_BYTES, stream>>>(xg, whhb, hbf, hf32, bar);

  k_final<<<1, 256, 0, stream>>>(hf32, Wout, bout, out);
}
